// VectorQuantizer_47055661695546
// MI455X (gfx1250) — matrix operand order repaired and checked by reading, not compiled, not run on hardware
//
#include <hip/hip_runtime.h>
#include <stdint.h>

#define N_PTS 100000
#define DIM   128
#define K_CB  2048

typedef __attribute__((ext_vector_type(16))) __bf16          v16bf;
typedef __attribute__((ext_vector_type(8)))  float           v8f;
typedef __attribute__((ext_vector_type(8)))  unsigned short  us8;
typedef __attribute__((ext_vector_type(4)))  unsigned short  us4;
typedef __attribute__((ext_vector_type(4)))  float           f32x4;

union FragAB { v16bf v; us8 h[2]; };

// round-to-nearest-even f32 -> bf16 bits; also returns the f32 value of that bf16
__device__ __forceinline__ unsigned short bf16_round(float x, float* as_f32) {
  union { float f; unsigned int u; } c;
  c.f = x;
  unsigned int r = (c.u + 0x7FFFu + ((c.u >> 16) & 1u)) & 0xFFFF0000u;
  c.u = r;
  *as_f32 = c.f;
  return (unsigned short)(r >> 16);
}

// ---------------------------------------------------------------------------
// Kernel A: codebook -> bf16 hi/lo split, squared norms, zero loss accumulator
// ---------------------------------------------------------------------------
__global__ void __launch_bounds__(DIM)
vq_prep(const float* __restrict__ cb,
        unsigned short* __restrict__ e_hi,
        unsigned short* __restrict__ e_lo,
        float* __restrict__ e_sq,
        float* __restrict__ accum) {
  const int k = blockIdx.x;
  const int d = threadIdx.x;
  const float v = cb[(size_t)k * DIM + d];
  float hif, dummy;
  const unsigned short hb = bf16_round(v, &hif);
  const unsigned short lb = bf16_round(v - hif, &dummy);
  e_hi[(size_t)k * DIM + d] = hb;
  e_lo[(size_t)k * DIM + d] = lb;

  __shared__ float red[DIM];
  red[d] = v * v;
  __syncthreads();
  #pragma unroll
  for (int s = DIM / 2; s > 0; s >>= 1) {
    if (d < s) red[d] += red[d + s];
    __syncthreads();
  }
  if (d == 0) e_sq[k] = red[0];
  if (k == 0 && d == 0) *accum = 0.0f;
}

// ---------------------------------------------------------------------------
// Kernel B: one wave32 per 32 rows of x. bf16-split WMMA GEMM + running argmin.
// score(n,k) = ||e_k||^2 - 2 * x_n . e_k   (x^2 term dropped: row-constant)
// ---------------------------------------------------------------------------
__global__ void __launch_bounds__(32)
vq_argmin(const float* __restrict__ x,
          const unsigned short* __restrict__ e_hi,
          const unsigned short* __restrict__ e_lo,
          const float* __restrict__ e_sq,
          int* __restrict__ idx_out,
          float* __restrict__ idx_f_out) {
  __shared__ unsigned short lds_hi[32][DIM];
  __shared__ unsigned short lds_lo[32][DIM];

  const int lane = threadIdx.x;          // 0..31
  const int row0 = blockIdx.x * 32;      // N_PTS % 32 == 0 -> no guards needed
  const int hf   = lane >> 4;            // half of the wave (0/1)
  const int lm   = lane & 15;

  // ---- stage 32x128 f32 tile of x as bf16 hi/lo in LDS (coalesced b128 loads)
  #pragma unroll 4
  for (int j = 0; j < 32; ++j) {
    const int e4 = j * 32 + lane;        // float4 index: 32 per row
    const int r  = e4 >> 5;
    const int c  = (e4 & 31) * 4;
    const f32x4 xv = *(const f32x4*)(x + (size_t)(row0 + r) * DIM + c);
    us4 hb, lb;
    #pragma unroll
    for (int q = 0; q < 4; ++q) {
      float hif, dummy;
      hb[q] = bf16_round(xv[q], &hif);
      lb[q] = bf16_round(xv[q] - hif, &dummy);
    }
    *(us4*)&lds_hi[r][c] = hb;
    *(us4*)&lds_lo[r][c] = lb;
  }
  __syncthreads();

  // ---- build resident A fragments: 2 row-tiles x 4 k-steps, hi & lo terms
  // A 16x32 bf16 layout: lane<16 holds K {d0..d0+7, d0+16..d0+23} with d0=0,
  // lanes 16..31 the same rows with d0=8.
  FragAB a_hi[2][4], a_lo[2][4];
  #pragma unroll
  for (int t = 0; t < 2; ++t) {
    const int r = t * 16 + lm;
    #pragma unroll
    for (int s = 0; s < 4; ++s) {
      const int d0 = s * 32 + hf * 8;
      a_hi[t][s].h[0] = *(const us8*)&lds_hi[r][d0];
      a_hi[t][s].h[1] = *(const us8*)&lds_hi[r][d0 + 16];
      a_lo[t][s].h[0] = *(const us8*)&lds_lo[r][d0];
      a_lo[t][s].h[1] = *(const us8*)&lds_lo[r][d0 + 16];
    }
  }

  float minval[2][8];
  int   minidx[2][8];
  #pragma unroll
  for (int t = 0; t < 2; ++t)
    #pragma unroll
    for (int r = 0; r < 8; ++r) { minval[t][r] = 3.4e38f; minidx[t][r] = 0; }

  // ---- sweep codebook in 16-column WMMA tiles
  for (int kt = 0; kt < K_CB / 16; ++kt) {
    const int   col   = kt * 16 + lm;    // this lane's codebook entry (C col)
    const float esq_c = e_sq[col];

    v8f acc0 = {0.f,0.f,0.f,0.f,0.f,0.f,0.f,0.f};
    v8f acc1 = {0.f,0.f,0.f,0.f,0.f,0.f,0.f,0.f};

    #pragma unroll
    for (int s = 0; s < 4; ++s) {
      // B 32x16 bf16 layout: lane = column, K {d0..d0+7, d0+16..d0+23} with
      // d0 = (lane<16 ? 0 : 8) within the 32-wide k-step (matches A).
      const size_t boff = (size_t)col * DIM + s * 32 + hf * 8;
      FragAB b_hi, b_lo;
      b_hi.h[0] = *(const us8*)(e_hi + boff);
      b_hi.h[1] = *(const us8*)(e_hi + boff + 16);
      b_lo.h[0] = *(const us8*)(e_lo + boff);
      b_lo.h[1] = *(const us8*)(e_lo + boff + 16);

      // dot ~= hi*hi + lo*hi + hi*lo  (lo*lo negligible)
      acc0 = __builtin_amdgcn_wmma_f32_16x16x32_bf16(false, a_hi[0][s].v, false, b_hi.v, (short)0, acc0, false, false);
      acc0 = __builtin_amdgcn_wmma_f32_16x16x32_bf16(false, a_lo[0][s].v, false, b_hi.v, (short)0, acc0, false, false);
      acc0 = __builtin_amdgcn_wmma_f32_16x16x32_bf16(false, a_hi[0][s].v, false, b_lo.v, (short)0, acc0, false, false);

      acc1 = __builtin_amdgcn_wmma_f32_16x16x32_bf16(false, a_hi[1][s].v, false, b_hi.v, (short)0, acc1, false, false);
      acc1 = __builtin_amdgcn_wmma_f32_16x16x32_bf16(false, a_lo[1][s].v, false, b_hi.v, (short)0, acc1, false, false);
      acc1 = __builtin_amdgcn_wmma_f32_16x16x32_bf16(false, a_hi[1][s].v, false, b_lo.v, (short)0, acc1, false, false);
    }

    // C layout: lane = column (lm), VGPR r -> row r (lanes<16) / row 8+r.
    // All 8 scores in this lane belong to codebook entry `col`.
    #pragma unroll
    for (int r = 0; r < 8; ++r) {
      const float sc0 = esq_c - 2.0f * acc0[r];
      if (sc0 < minval[0][r]) { minval[0][r] = sc0; minidx[0][r] = col; }
      const float sc1 = esq_c - 2.0f * acc1[r];
      if (sc1 < minval[1][r]) { minval[1][r] = sc1; minidx[1][r] = col; }
    }
  }

  // ---- reduce across the 16 columns held by each wave-half; prefer lower idx
  #pragma unroll
  for (int t = 0; t < 2; ++t) {
    #pragma unroll
    for (int r = 0; r < 8; ++r) {
      float v = minval[t][r];
      int   i = minidx[t][r];
      #pragma unroll
      for (int m = 8; m >= 1; m >>= 1) {
        const float ov = __shfl_xor(v, m, 32);
        const int   oi = __shfl_xor(i, m, 32);
        if (ov < v || (ov == v && oi < i)) { v = ov; i = oi; }
      }
      if (lm == 0) {
        const int row = row0 + t * 16 + hf * 8 + r;
        idx_out[row]   = i;
        idx_f_out[row] = (float)i;
      }
    }
  }
}

// ---------------------------------------------------------------------------
// Kernel C: gather quantized rows (== quantized_st numerically) + MSE partials
// ---------------------------------------------------------------------------
__global__ void __launch_bounds__(DIM)
vq_gather_loss(const float* __restrict__ x,
               const float* __restrict__ cb,
               const int* __restrict__ idx,
               float* __restrict__ q_out,
               float* __restrict__ accum) {
  const int n = blockIdx.x;
  const int d = threadIdx.x;
  const int k = idx[n];
  const float q = cb[(size_t)k * DIM + d];
  const float h = x[(size_t)n * DIM + d];
  q_out[(size_t)n * DIM + d] = q;
  const float diff = q - h;
  float v = diff * diff;
  #pragma unroll
  for (int m = 16; m >= 1; m >>= 1) v += __shfl_xor(v, m, 32);
  if ((d & 31) == 0) atomicAdd(accum, v);
}

// ---------------------------------------------------------------------------
// Kernel D: total_loss = (1 + 0.25) * mean((q-h)^2)
// ---------------------------------------------------------------------------
__global__ void vq_finalize(const float* __restrict__ accum,
                            float* __restrict__ loss_out) {
  *loss_out = *accum * (1.25f / ((float)N_PTS * (float)DIM));
}

// ---------------------------------------------------------------------------
extern "C" void kernel_launch(void* const* d_in, const int* in_sizes, int n_in,
                              void* d_out, int out_size, void* d_ws, size_t ws_size,
                              hipStream_t stream) {
  const float* x  = (const float*)d_in[0];   // [100000,128] f32
  const float* cb = (const float*)d_in[1];   // [2048,128]   f32

  float* out   = (float*)d_out;
  float* q_out = out;                                  // 12,800,000
  float* idx_f = out + (size_t)N_PTS * DIM;            //    100,000
  float* loss  = out + (size_t)N_PTS * DIM + N_PTS;    //          1

  char* ws = (char*)d_ws;
  unsigned short* e_hi = (unsigned short*)(ws);                       // 512 KB
  unsigned short* e_lo = (unsigned short*)(ws + 524288);              // 512 KB
  float* e_sq  = (float*)(ws + 1048576);                              //   8 KB
  int*   idx   = (int*)  (ws + 1048576 + 8192);                       // 400 KB
  float* accum = (float*)(ws + 1048576 + 8192 + 400000);              //   4 B

  vq_prep       <<<K_CB,       DIM, 0, stream>>>(cb, e_hi, e_lo, e_sq, accum);
  vq_argmin     <<<N_PTS / 32, 32,  0, stream>>>(x, e_hi, e_lo, e_sq, idx, idx_f);
  vq_gather_loss<<<N_PTS,      DIM, 0, stream>>>(x, cb, idx, q_out, accum);
  vq_finalize   <<<1,          1,   0, stream>>>(accum, loss);
}
